// GCN_BiLSTM_Model_48275432407204
// MI455X (gfx1250) — matrix operand order repaired and checked by reading, not compiled, not run on hardware
//
#include <hip/hip_runtime.h>

// ---------------- CDNA5 WMMA types ----------------
typedef __attribute__((ext_vector_type(16))) __bf16 bf16x16;
typedef __attribute__((ext_vector_type(4)))  __bf16 bf16x4;
typedef __attribute__((ext_vector_type(8)))  float  f32x8;

union FragBF { uint4 q[2]; bf16x16 v; };

#define BM 128
#define BN 128
#define BK 32

// ============================================================
// Generic bf16-WMMA GEMM:  Out[M][N] = epi( A[M][Ktot] * Wt[N][Ktot]^T )
// A = concat_K( A1 (f32 or bf16 per template), A2 (f32) ), K1 % 32 == 0.
// grid.z replicates the problem (forward/backward LSTM directions).
// Block tile 128x128, 8 waves (4M x 2N), 32x64 per wave = 2x4 WMMA frags.
// LDS double-buffered; global->reg prefetch overlaps the WMMA chain.
// ============================================================
template<int A1BF, int EPI, int RAW>
__global__ __launch_bounds__(256) void wmma_gemm(
    const void* __restrict__ A1, long ldA1, long offA1, long zstrA1,
    const float* __restrict__ A2, long ldA2, long zstrA2,
    int K1, int Ktot,
    const __bf16* __restrict__ Wt, long zstrW,
    const float* __restrict__ bias, long zstrBias,
    const float* __restrict__ bng, const float* __restrict__ bnb,
    const float* __restrict__ bnm, const float* __restrict__ bnv,
    float* __restrict__ Out, long ldOut, long zstrOut, int NStore,
    float* __restrict__ raw75)
{
    __shared__ __align__(16) __bf16 As[2][BM][BK];
    __shared__ __align__(16) __bf16 Bs[2][BN][BK];

    const int z    = blockIdx.z;
    const int tid  = threadIdx.x;
    const int lane = tid & 31;
    const int wave = tid >> 5;          // 0..7
    const int wm   = wave & 3;          // M sub-tile (x32)
    const int wn   = wave >> 2;         // N sub-tile (x64)
    const int lo   = lane & 15;
    const int hi   = lane >> 4;

    const __bf16* W = Wt + (long)z * zstrW;
    const long rowBlock = (long)blockIdx.x * BM;
    const long colBlock = (long)blockIdx.y * BN;

    // staging thread roles
    const int tr  = tid >> 3;           // 0..31  (A row within pass)
    const int tk  = (tid & 7) * 4;      // 0..28  (A k, 4 elements)
    const int tnB = tid >> 1;           // 0..127 (B row)
    const int tkB = (tid & 1) * 16;     // 0/16   (B k, 16 elements)

    const long a1base = offA1 + zstrA1 * z;
    const long a2base = zstrA2 * z;

    auto fetchA = [&](int kb, bf16x4 (&dst)[4]) {
        if (kb < K1) {
            #pragma unroll
            for (int p = 0; p < 4; ++p) {
                const long eo = a1base + (rowBlock + p * 32 + tr) * ldA1 + (kb + tk);
                if (A1BF) {
                    dst[p] = *(const bf16x4*)((const __bf16*)A1 + eo);
                } else {
                    const float4 f = *(const float4*)((const float*)A1 + eo);
                    dst[p] = (bf16x4){(__bf16)f.x, (__bf16)f.y, (__bf16)f.z, (__bf16)f.w};
                }
            }
        } else {
            #pragma unroll
            for (int p = 0; p < 4; ++p) {
                const long eo = a2base + (rowBlock + p * 32 + tr) * ldA2 + (kb + tk - K1);
                const float4 f = *(const float4*)(A2 + eo);
                dst[p] = (bf16x4){(__bf16)f.x, (__bf16)f.y, (__bf16)f.z, (__bf16)f.w};
            }
        }
    };
    auto fetchB = [&](int kb, uint4 (&dst)[2]) {
        const __bf16* src = W + (colBlock + tnB) * (long)Ktot + kb + tkB;
        dst[0] = *(const uint4*)src;
        dst[1] = *(const uint4*)(src + 8);
    };
    auto stage = [&](int buf, const bf16x4 (&ra)[4], const uint4 (&rb)[2]) {
        #pragma unroll
        for (int p = 0; p < 4; ++p)
            *(bf16x4*)&As[buf][p * 32 + tr][tk] = ra[p];
        *(uint4*)&Bs[buf][tnB][tkB]     = rb[0];
        *(uint4*)&Bs[buf][tnB][tkB + 8] = rb[1];
    };

    f32x8 acc[2][4] = {};
    bf16x4 ra[4];
    uint4  rb[2];

    fetchA(0, ra);
    fetchB(0, rb);
    stage(0, ra, rb);
    __syncthreads();

    const int nk = Ktot / BK;
    for (int kb = 0; kb < nk; ++kb) {
        const int cur = kb & 1;
        if (kb + 1 < nk) {                    // prefetch next tile into registers
            fetchA((kb + 1) * BK, ra);
            fetchB((kb + 1) * BK, rb);
        }
        // ---- fragment loads (ISA 7.12.2 layouts) ----
        FragBF a[2], b[4];
        #pragma unroll
        for (int s = 0; s < 2; ++s) {
            const __bf16* row = &As[cur][wm * 32 + s * 16 + lo][0];
            a[s].q[0] = *(const uint4*)(row + hi * 8);        // K = hi*8 .. +7
            a[s].q[1] = *(const uint4*)(row + 16 + hi * 8);   // K = 16+hi*8 .. +7
        }
        #pragma unroll
        for (int u = 0; u < 4; ++u) {
            const __bf16* row = &Bs[cur][wn * 64 + u * 16 + lo][0];
            b[u].q[0] = *(const uint4*)(row + hi * 8);        // K = hi*8 .. +7
            b[u].q[1] = *(const uint4*)(row + 16 + hi * 8);   // K = 16+hi*8 .. +7
        }
        #pragma unroll
        for (int s = 0; s < 2; ++s)
            #pragma unroll
            for (int u = 0; u < 4; ++u)
                acc[s][u] = __builtin_amdgcn_wmma_f32_16x16x32_bf16(
                    false, a[s].v, false, b[u].v, (short)0, acc[s][u], false, false);

        if (kb + 1 < nk) stage(cur ^ 1, ra, rb);  // write the other buffer
        __syncthreads();
    }

    // ---- epilogue: raw dump (rows<75), bias, ReLU, BN, store ----
    #pragma unroll
    for (int u = 0; u < 4; ++u) {
        const int gcol = (int)colBlock + wn * 64 + u * 16 + lo;
        if (gcol >= NStore) continue;
        const float bv = bias[zstrBias * z + gcol];
        float scale = 0.f, shift = 0.f;
        if (EPI == 2) {
            scale = rsqrtf(bnv[gcol] + 1e-5f) * bng[gcol];
            shift = bnb[gcol] - bnm[gcol] * scale;
        }
        #pragma unroll
        for (int s = 0; s < 2; ++s) {
            #pragma unroll
            for (int r = 0; r < 8; ++r) {
                const long grow = rowBlock + wm * 32 + s * 16 + 8 * hi + r;
                float val = acc[s][u][r];
                if (RAW) { if (grow < 75) raw75[grow * ldOut + gcol] = val; }
                val += bv;
                if (EPI >= 1) val = fmaxf(val, 0.f);
                if (EPI == 2) val = val * scale + shift;
                Out[zstrOut * z + grow * ldOut + gcol] = val;
            }
        }
    }
}

// ============================================================
// GCN layer 1 (K=3): trivial dense part, dumps raw hW for rows<75
// ============================================================
__global__ __launch_bounds__(256) void gcn_layer1(
    const float* __restrict__ x, const float* __restrict__ W1, const float* __restrict__ b1,
    const float* __restrict__ bng, const float* __restrict__ bnb,
    const float* __restrict__ bnm, const float* __restrict__ bnv,
    float* __restrict__ out, float* __restrict__ raw)
{
    const long idx = (long)blockIdx.x * 256 + threadIdx.x;   // row*256 + n
    const int  n   = (int)(idx & 255);
    const long row = idx >> 8;
    const float x0 = x[row * 3], x1 = x[row * 3 + 1], x2 = x[row * 3 + 2];
    float v = x0 * W1[n] + x1 * W1[256 + n] + x2 * W1[512 + n];
    if (row < 75) raw[row * 256 + n] = v;
    v += b1[n];
    v = fmaxf(v, 0.f);
    v = (v - bnm[n]) * rsqrtf(bnv[n] + 1e-5f) * bng[n] + bnb[n];
    out[idx] = v;
}

// ============================================================
// GCN fixup: redo rows 0..74 with symmetric-norm edge aggregation
// ============================================================
__global__ __launch_bounds__(256) void gcn_fixup(
    const float* __restrict__ raw, const int* __restrict__ edges,
    const float* __restrict__ bias, int epimode,
    const float* __restrict__ bng, const float* __restrict__ bnb,
    const float* __restrict__ bnm, const float* __restrict__ bnv,
    float* __restrict__ out)
{
    __shared__ int   es[116], ed[116];
    __shared__ float deg[75], dis[75];
    const int t = threadIdx.x;
    if (t < 75) deg[t] = 1.f;
    if (t < 116) { es[t] = edges[t]; ed[t] = edges[116 + t]; }
    __syncthreads();
    if (t < 116) atomicAdd(&deg[ed[t]], 1.f);
    __syncthreads();
    if (t < 75) dis[t] = rsqrtf(deg[t]);
    __syncthreads();
    const int n = t;   // column 0..255
    for (int i = 0; i < 75; ++i) {
        float a = raw[i * 256 + n] / deg[i];
        for (int e = 0; e < 116; ++e)
            if (ed[e] == i) a += dis[es[e]] * dis[i] * raw[es[e] * 256 + n];
        float v = a + bias[n];
        if (epimode >= 1) v = fmaxf(v, 0.f);
        if (epimode == 2) v = (v - bnm[n]) * rsqrtf(bnv[n] + 1e-5f) * bng[n] + bnb[n];
        out[i * 256 + n] = v;
    }
}

// ============================================================
// LSTM gate update (both directions): c,h from gates G
// ============================================================
__device__ inline float sigf(float x) { return 1.f / (1.f + __expf(-x)); }

__global__ __launch_bounds__(256) void lstm_gate(
    const float* __restrict__ G,          // [2][2048][1024]
    float* __restrict__ hst, float* __restrict__ cst,   // [2][2048][256]
    __bf16* __restrict__ hs,              // [75][2048][512] bf16 or null
    int t_fwd, int t_bwd,
    float* __restrict__ last,             // [2048][512] or null
    int cap_fwd, int cap_bwd)
{
    const long idx = (long)blockIdx.x * 256 + threadIdx.x;  // (d*2048+b)*256 + j
    const int  j   = (int)(idx & 255);
    const long bd  = idx >> 8;
    const int  b   = (int)(bd & 2047);
    const int  d   = (int)(bd >> 11);
    const float* g = G + bd * 1024;
    const float gi = sigf(g[j]);
    const float gf = sigf(g[j + 256]);
    const float gg = tanhf(g[j + 512]);
    const float go = sigf(g[j + 768]);
    const float c  = gf * cst[idx] + gi * gg;
    const float h  = go * tanhf(c);
    cst[idx] = c;
    hst[idx] = h;
    if (hs) {
        const int t = d ? t_bwd : t_fwd;
        hs[((long)t * 2048 + b) * 512 + d * 256 + j] = (__bf16)h;
    }
    if (last && ((d == 0 && cap_fwd) || (d == 1 && cap_bwd)))
        last[(long)b * 512 + d * 256 + j] = h;
}

// ============================================================
// Prep kernels (weight layout conversion to bf16 [N][K])
// ============================================================
__global__ void prep_transpose_bf16(const float* __restrict__ W, __bf16* __restrict__ Wt,
                                    int K, int N) {
    const long idx = (long)blockIdx.x * 256 + threadIdx.x;
    if (idx >= (long)N * K) return;
    const int  k = (int)(idx % K);
    const long n = idx / K;
    Wt[idx] = (__bf16)W[(long)k * N + n];
}
__global__ void prep_concat_bf16(const float* __restrict__ Wa, const float* __restrict__ Wb,
                                 __bf16* __restrict__ Wt, int N, int K1, int K2) {
    const long idx = (long)blockIdx.x * 256 + threadIdx.x;
    const int Kt = K1 + K2;
    if (idx >= 2L * N * Kt) return;
    const int  k  = (int)(idx % Kt);
    const long nd = idx / Kt;
    const int  n  = (int)(nd % N);
    const int  d  = (int)(nd / N);
    const float v = (k < K1) ? Wa[((long)d * N + n) * K1 + k]
                             : Wb[((long)d * N + n) * K2 + (k - K1)];
    Wt[idx] = (__bf16)v;
}
__global__ void prep_bias_sum(const float* a, const float* b, float* o, int n) {
    const int i = blockIdx.x * 256 + threadIdx.x;
    if (i < n) o[i] = a[i] + b[i];
}
__global__ void prep_fc(const float* __restrict__ fcW, __bf16* __restrict__ Wt) {
    const long idx = (long)blockIdx.x * 256 + threadIdx.x;   // 256*512
    if (idx >= 256L * 512) return;
    const int k = (int)(idx & 511);
    const int n = (int)(idx >> 9);
    Wt[idx] = (n < 250) ? (__bf16)fcW[(long)n * 512 + k] : (__bf16)0.f;
}
__global__ void zero_f32(float* p, long n) {
    const long i = (long)blockIdx.x * 256 + threadIdx.x;
    if (i < n) p[i] = 0.f;
}

// ============================================================
// Host orchestration
// ============================================================
template<int A1BF, int EPI, int RAW>
static void launch_gemm(hipStream_t stream,
                        const void* A1, long ldA1, long offA1, long zsA1,
                        const float* A2, long ldA2, long zsA2, int K1, int Kt,
                        const __bf16* Wt, long zsW, const float* bias, long zsB,
                        const float* g, const float* bb, const float* m, const float* v,
                        float* Out, long ldOut, long zsOut, int NStore, float* rawp,
                        int Mm, int Nn, int Z) {
    dim3 gr((unsigned)(Mm / BM), (unsigned)((Nn + BN - 1) / BN), (unsigned)Z);
    wmma_gemm<A1BF, EPI, RAW><<<gr, 256, 0, stream>>>(
        A1, ldA1, offA1, zsA1, A2, ldA2, zsA2, K1, Kt, Wt, zsW, bias, zsB,
        g, bb, m, v, Out, ldOut, zsOut, NStore, rawp);
}

extern "C" void kernel_launch(void* const* d_in, const int* in_sizes, int n_in,
                              void* d_out, int out_size, void* d_ws, size_t ws_size,
                              hipStream_t stream) {
    (void)in_sizes; (void)n_in; (void)out_size; (void)ws_size;
    const float* x    = (const float*)d_in[0];
    const float* W1   = (const float*)d_in[1];
    const float* b1   = (const float*)d_in[2];
    const float* W2   = (const float*)d_in[3];
    const float* b2   = (const float*)d_in[4];
    const float* W3   = (const float*)d_in[5];
    const float* b3   = (const float*)d_in[6];
    const float* W4   = (const float*)d_in[7];
    const float* b4   = (const float*)d_in[8];
    const float* bn1g = (const float*)d_in[9],  *bn1b = (const float*)d_in[10];
    const float* bn1m = (const float*)d_in[11], *bn1v = (const float*)d_in[12];
    const float* bn2g = (const float*)d_in[13], *bn2b = (const float*)d_in[14];
    const float* bn2m = (const float*)d_in[15], *bn2v = (const float*)d_in[16];
    const float* Wih0 = (const float*)d_in[17], *Whh0 = (const float*)d_in[18];
    const float* bih0 = (const float*)d_in[19], *bhh0 = (const float*)d_in[20];
    const float* Wih1 = (const float*)d_in[21], *Whh1 = (const float*)d_in[22];
    const float* bih1 = (const float*)d_in[23], *bhh1 = (const float*)d_in[24];
    const float* fcW  = (const float*)d_in[25], *fcb  = (const float*)d_in[26];
    const int*   edge = (const int*)d_in[27];

    const long M = 153600;   // 2048*75

    char* w = (char*)d_ws;
    auto alloc = [&](size_t bytes) { void* p = (void*)w; w += (bytes + 255) & ~(size_t)255; return p; };
    float*  hA    = (float*) alloc(M * 256 * 4);
    float*  hB    = (float*) alloc(M * 256 * 4);
    float*  raw   = (float*) alloc(75 * 256 * 4);
    __bf16* W2t   = (__bf16*)alloc(256 * 256 * 2);
    __bf16* W3t   = (__bf16*)alloc(256 * 256 * 2);
    __bf16* W4t   = (__bf16*)alloc(256 * 256 * 2);
    __bf16* Wc0   = (__bf16*)alloc(2L * 1024 * 512 * 2);
    __bf16* Wc1   = (__bf16*)alloc(2L * 1024 * 768 * 2);
    float*  bs0   = (float*) alloc(2048 * 4);
    float*  bs1   = (float*) alloc(2048 * 4);
    __bf16* fcWt  = (__bf16*)alloc(256L * 512 * 2);
    __bf16* hs0   = (__bf16*)alloc(75L * 2048 * 512 * 2);
    float*  G     = (float*) alloc(2L * 2048 * 1024 * 4);
    float*  hst   = (float*) alloc(2L * 2048 * 256 * 4);
    float*  cst   = (float*) alloc(2L * 2048 * 256 * 4);
    float*  last  = (float*) alloc(2048L * 512 * 4);

    // ---- weight prep ----
    prep_transpose_bf16<<<256, 256, 0, stream>>>(W2, W2t, 256, 256);
    prep_transpose_bf16<<<256, 256, 0, stream>>>(W3, W3t, 256, 256);
    prep_transpose_bf16<<<256, 256, 0, stream>>>(W4, W4t, 256, 256);
    prep_concat_bf16<<<4096, 256, 0, stream>>>(Wih0, Whh0, Wc0, 1024, 256, 256);
    prep_concat_bf16<<<6144, 256, 0, stream>>>(Wih1, Whh1, Wc1, 1024, 512, 256);
    prep_bias_sum<<<8, 256, 0, stream>>>(bih0, bhh0, bs0, 2048);
    prep_bias_sum<<<8, 256, 0, stream>>>(bih1, bhh1, bs1, 2048);
    prep_fc<<<512, 256, 0, stream>>>(fcW, fcWt);

    // ---- GCN ----
    gcn_layer1<<<(unsigned)(M), 256, 0, stream>>>(x, W1, b1, bn1g, bn1b, bn1m, bn1v, hA, raw);
    gcn_fixup<<<1, 256, 0, stream>>>(raw, edge, b1, 2, bn1g, bn1b, bn1m, bn1v, hA);

    launch_gemm<0, 2, 1>(stream, hA, 256, 0, 0, nullptr, 0, 0, 256, 256,
                         W2t, 0, b2, 0, bn2g, bn2b, bn2m, bn2v,
                         hB, 256, 0, 256, raw, (int)M, 256, 1);
    gcn_fixup<<<1, 256, 0, stream>>>(raw, edge, b2, 2, bn2g, bn2b, bn2m, bn2v, hB);

    launch_gemm<0, 1, 1>(stream, hB, 256, 0, 0, nullptr, 0, 0, 256, 256,
                         W3t, 0, b3, 0, nullptr, nullptr, nullptr, nullptr,
                         hA, 256, 0, 256, raw, (int)M, 256, 1);
    gcn_fixup<<<1, 256, 0, stream>>>(raw, edge, b3, 1, nullptr, nullptr, nullptr, nullptr, hA);

    launch_gemm<0, 0, 1>(stream, hA, 256, 0, 0, nullptr, 0, 0, 256, 256,
                         W4t, 0, b4, 0, nullptr, nullptr, nullptr, nullptr,
                         hB, 256, 0, 256, raw, (int)M, 256, 1);
    gcn_fixup<<<1, 256, 0, stream>>>(raw, edge, b4, 0, nullptr, nullptr, nullptr, nullptr, hB);
    // hB = h4, xs[t][b] = hB[b*75+t]

    // ---- BiLSTM layer 0 ----
    zero_f32<<<4096, 256, 0, stream>>>(hst, 2L * 2048 * 256);
    zero_f32<<<4096, 256, 0, stream>>>(cst, 2L * 2048 * 256);
    for (int s = 0; s < 75; ++s) {
        const long tf = s, tb = 74 - s;
        launch_gemm<0, 0, 0>(stream, hB, 75L * 256, tf * 256, (tb - tf) * 256,
                             hst, 256, 2048L * 256, 256, 512,
                             Wc0, 1024L * 512, bs0, 1024,
                             nullptr, nullptr, nullptr, nullptr,
                             G, 1024, 2048L * 1024, 1024, nullptr, 2048, 1024, 2);
        lstm_gate<<<4096, 256, 0, stream>>>(G, hst, cst, hs0, (int)tf, (int)tb,
                                            nullptr, 0, 0);
    }

    // ---- BiLSTM layer 1 ----
    zero_f32<<<4096, 256, 0, stream>>>(hst, 2L * 2048 * 256);
    zero_f32<<<4096, 256, 0, stream>>>(cst, 2L * 2048 * 256);
    for (int s = 0; s < 75; ++s) {
        const long tf = s, tb = 74 - s;
        launch_gemm<1, 0, 0>(stream, hs0, 512, tf * 2048 * 512, (tb - tf) * 2048 * 512,
                             hst, 256, 2048L * 256, 512, 768,
                             Wc1, 1024L * 768, bs1, 1024,
                             nullptr, nullptr, nullptr, nullptr,
                             G, 1024, 2048L * 1024, 1024, nullptr, 2048, 1024, 2);
        lstm_gate<<<4096, 256, 0, stream>>>(G, hst, cst, nullptr, 0, 0,
                                            last, s == 74, s == 0);
    }

    // ---- final FC: [2048,512] @ fcW^T + fcb -> d_out [2048,250] ----
    launch_gemm<0, 0, 0>(stream, last, 512, 0, 0, nullptr, 0, 0, 512, 512,
                         fcWt, 0, fcb, 0, nullptr, nullptr, nullptr, nullptr,
                         (float*)d_out, 250, 0, 250, nullptr, 2048, 256, 1);
}
